// PointNetClassifier_32083405701129
// MI455X (gfx1250) — matrix operand order repaired and checked by reading, not compiled, not run on hardware
//
#include <hip/hip_runtime.h>
#include <hip/hip_bf16.h>
#include <cstdint>

typedef _Float16 f16;
typedef __attribute__((ext_vector_type(8)))  _Float16 v8h;
typedef __attribute__((ext_vector_type(16))) _Float16 v16h;
typedef __attribute__((ext_vector_type(8)))  float    v8f;

#define NB      16
#define DIM     96
#define DHW     (96*96*96)      // 884736 voxels per batch
#define NPTS    4096
#define CHUNK   4096
#define NCHUNK  (DHW/CHUNK)     // 216 chunks per batch
#define THRESH  0.5f

// ---------------------------------------------------------------------------
// 1) Count foreground voxels per (batch, chunk)
// ---------------------------------------------------------------------------
__global__ void pn_count(const float* __restrict__ vol, int* __restrict__ counts) {
    __shared__ int red[256];
    const int b = blockIdx.y, chunk = blockIdx.x;
    const float* v = vol + (size_t)b * DHW + (size_t)chunk * CHUNK;
    int c = 0;
    for (int i = threadIdx.x; i < CHUNK; i += 256) c += (v[i] > THRESH) ? 1 : 0;
    red[threadIdx.x] = c;
    __syncthreads();
    for (int s = 128; s > 0; s >>= 1) {
        if (threadIdx.x < s) red[threadIdx.x] += red[threadIdx.x + s];
        __syncthreads();
    }
    if (threadIdx.x == 0) counts[b * NCHUNK + chunk] = red[0];
}

// ---------------------------------------------------------------------------
// 2) Per-batch exclusive scan of chunk counts (one 256-thread block per batch)
// ---------------------------------------------------------------------------
__global__ void pn_scan(const int* __restrict__ counts, int* __restrict__ offs,
                        int* __restrict__ totals) {
    __shared__ int s[256];
    const int b = blockIdx.x, t = threadIdx.x;
    const int v = (t < NCHUNK) ? counts[b * NCHUNK + t] : 0;
    s[t] = v;
    __syncthreads();
    for (int d = 1; d < 256; d <<= 1) {
        int add = (t >= d) ? s[t - d] : 0;
        __syncthreads();
        s[t] += add;
        __syncthreads();
    }
    if (t < NCHUNK) offs[b * NCHUNK + t] = s[t] - v;   // exclusive
    if (t == 255)   totals[b] = s[255];
}

// ---------------------------------------------------------------------------
// 3) Ordered scatter of foreground flat indices (stable, row-major)
// ---------------------------------------------------------------------------
__global__ void pn_scatter(const float* __restrict__ vol, const int* __restrict__ offs,
                           int* __restrict__ compact) {
    __shared__ int s[256];
    const int b = blockIdx.y, chunk = blockIdx.x;
    const float* v = vol + (size_t)b * DHW + (size_t)chunk * CHUNK;
    int* __restrict__ cmp = compact + (size_t)b * DHW;
    int run = offs[b * NCHUNK + chunk];               // uniform across block
    for (int tile = 0; tile < CHUNK; tile += 256) {
        const int i = tile + threadIdx.x;
        const int pred = (v[i] > THRESH) ? 1 : 0;
        s[threadIdx.x] = pred;
        __syncthreads();
        for (int d = 1; d < 256; d <<= 1) {           // inclusive scan
            int add = (threadIdx.x >= d) ? s[threadIdx.x - d] : 0;
            __syncthreads();
            s[threadIdx.x] += add;
            __syncthreads();
        }
        const int incl = s[threadIdx.x];
        const int tileTotal = s[255];
        if (pred) cmp[run + incl - 1] = chunk * CHUNK + i;
        __syncthreads();                              // protect s before next tile
        run += tileTotal;
    }
}

// ---------------------------------------------------------------------------
// 4) Sample 4096 points per batch -> X0[(b*4096+p)][32] f16 (3 coords + pad)
// ---------------------------------------------------------------------------
__global__ void pn_sample(const int* __restrict__ compact, const int* __restrict__ totals,
                          f16* __restrict__ X0) {
    const int b = blockIdx.y;
    const int p = blockIdx.x * 256 + threadIdx.x;
    const int n = totals[b];
    float px = 0.f, py = 0.f, pz = 0.f;
    if (n > 0) {
        int idx;
        if (n >= NPTS) {
            int lin = (int)((float)p * (float)(n - 1) / (float)(NPTS - 1));
            lin = lin < 0 ? 0 : (lin > n - 1 ? n - 1 : lin);
            idx = lin;
        } else {
            idx = p % n;
        }
        const int flat = compact[(size_t)b * DHW + idx];
        const int dd = flat / (DIM * DIM);
        const int hh = (flat / DIM) % DIM;
        const int ww = flat % DIM;
        const float sc = 2.0f / (float)(DIM - 1);
        px = (float)dd * sc - 1.0f;
        py = (float)hh * sc - 1.0f;
        pz = (float)ww * sc - 1.0f;
    }
    f16* row = X0 + ((size_t)b * NPTS + p) * 32;
    row[0] = (f16)px; row[1] = (f16)py; row[2] = (f16)pz;
    #pragma unroll
    for (int c = 3; c < 32; ++c) row[c] = (f16)0.0f;
}

// ---------------------------------------------------------------------------
// 5) Weight -> WMMA B-fragment layout.  Fragment = 512 f16 (32 lanes x 16).
//    Lane L holds col n = n0 + (L&15), k = k0 + (L>=16 ? 8 : 0) + (e&7) + (e>=8 ? 16 : 0), e=0..15.
// ---------------------------------------------------------------------------
__global__ void pn_prep_wfrag(const float* __restrict__ W, f16* __restrict__ Wfrag,
                              int CinPad, int CinReal, int Cout) {
    const int idx = blockIdx.x * 256 + threadIdx.x;
    const int total = CinPad * Cout;
    if (idx >= total) return;
    const int frag = idx >> 9;
    const int lane = (idx >> 4) & 31;
    const int e    = idx & 15;
    const int ntiles = Cout >> 4;
    const int kt = frag / ntiles, nt = frag % ntiles;
    const int n = nt * 16 + (lane & 15);
    const int k = kt * 32 + ((lane >> 4) ? 8 : 0) + (e & 7) + ((e >> 3) ? 16 : 0);
    const float v = (k < CinReal) ? W[(size_t)n * CinReal + k] : 0.0f;
    Wfrag[idx] = (f16)v;
}

// ---------------------------------------------------------------------------
// 6) WMMA GEMM: Y[M,Cout] = X[M,CinPad] * Wfrag.  128 threads = 4 waves,
//    workgroup tile 64(M) x 64(N); each wave: 16(M) x 64(N) = 4 accumulators.
// ---------------------------------------------------------------------------
__launch_bounds__(128)
__global__ void pn_gemm_wmma(const f16* __restrict__ X, const f16* __restrict__ Wfrag,
                             f16* __restrict__ Y, int Cin, int Cout) {
    const int lane = threadIdx.x & 31;
    const int wave = threadIdx.x >> 5;
    const int lrow = lane & 15;
    const int half = lane >> 4;
    const int m0   = blockIdx.x * 64 + wave * 16;
    const int n0   = blockIdx.y * 64;
    const int ntiles = Cout >> 4;

    v8f acc0 = {}, acc1 = {}, acc2 = {}, acc3 = {};
    const f16* __restrict__ xrow = X + (size_t)(m0 + lrow) * Cin + half * 8;
    const f16* __restrict__ bcol = Wfrag + (size_t)(n0 >> 4) * 512 + lane * 16;

    for (int k0 = 0; k0 < Cin; k0 += 32) {
        __builtin_prefetch(xrow + k0 + 32, 0, 1);      // global_prefetch_b8
        // A fragment: 16-bit A 16x32 layout — lanes 0-15 hold K{0-7,16-23},
        // lanes 16-31 hold K{8-15,24-31}; two contiguous 16B loads.
        const v8h alo = *(const v8h*)(xrow + k0);
        const v8h ahi = *(const v8h*)(xrow + k0 + 16);
        const v16h a = __builtin_shufflevector(alo, ahi,
                          0,1,2,3,4,5,6,7,8,9,10,11,12,13,14,15);
        const f16* bp = bcol + (size_t)(k0 >> 5) * ntiles * 512;
        const v16h b0 = *(const v16h*)(bp);
        const v16h b1 = *(const v16h*)(bp + 512);
        const v16h b2 = *(const v16h*)(bp + 1024);
        const v16h b3 = *(const v16h*)(bp + 1536);
        acc0 = __builtin_amdgcn_wmma_f32_16x16x32_f16(false, a, false, b0, (short)0, acc0, false, false);
        acc1 = __builtin_amdgcn_wmma_f32_16x16x32_f16(false, a, false, b1, (short)0, acc1, false, false);
        acc2 = __builtin_amdgcn_wmma_f32_16x16x32_f16(false, a, false, b2, (short)0, acc2, false, false);
        acc3 = __builtin_amdgcn_wmma_f32_16x16x32_f16(false, a, false, b3, (short)0, acc3, false, false);
    }
    // 32-bit C/D 16x16: VGPR r, lane L -> row = r + (L<16?0:8), col = L&15
    const int mrow = m0 + half * 8;
    f16* __restrict__ yout = Y + (size_t)mrow * Cout + n0 + lrow;
    #pragma unroll
    for (int r = 0; r < 8; ++r) {
        f16* yr = yout + (size_t)r * Cout;
        yr[0]  = (f16)acc0[r];
        yr[16] = (f16)acc1[r];
        yr[32] = (f16)acc2[r];
        yr[48] = (f16)acc3[r];
    }
}

// ---------------------------------------------------------------------------
// 7) GroupNorm stats -> per-(b,channel) scale/shift.  Block per (b, group).
// ---------------------------------------------------------------------------
__global__ void pn_stats(const f16* __restrict__ Y, const float* __restrict__ gamma,
                         const float* __restrict__ beta, float* __restrict__ scale,
                         float* __restrict__ shift, int Cout) {
    __shared__ float ssum[256], ssq[256];
    __shared__ float s_mu, s_rs;
    const int b = blockIdx.x, g = blockIdx.y;
    const int cpg = Cout >> 3;                 // channels per group (pow2)
    const int lc  = __builtin_ctz(cpg);
    const f16* base = Y + (size_t)b * NPTS * Cout + g * cpg;
    const int total = NPTS * cpg;
    float sum = 0.f, sq = 0.f;
    for (int e = threadIdx.x; e < total; e += 256) {
        const int p = e >> lc, c = e & (cpg - 1);
        const float v = (float)base[(size_t)p * Cout + c];
        sum += v; sq += v * v;
    }
    ssum[threadIdx.x] = sum; ssq[threadIdx.x] = sq;
    __syncthreads();
    for (int s = 128; s > 0; s >>= 1) {
        if (threadIdx.x < s) {
            ssum[threadIdx.x] += ssum[threadIdx.x + s];
            ssq[threadIdx.x]  += ssq[threadIdx.x + s];
        }
        __syncthreads();
    }
    if (threadIdx.x == 0) {
        const float inv = 1.0f / (float)total;
        const float mu  = ssum[0] * inv;
        const float var = ssq[0] * inv - mu * mu;
        s_mu = mu;
        s_rs = rsqrtf(var + 1e-5f);
    }
    __syncthreads();
    for (int c = threadIdx.x; c < cpg; c += 256) {
        const int ch = g * cpg + c;
        const float sc = s_rs * gamma[ch];
        scale[b * Cout + ch] = sc;
        shift[b * Cout + ch] = beta[ch] - s_mu * sc;
    }
}

// ---------------------------------------------------------------------------
// 8) Elementwise relu(scale*y + shift) -> next-layer f16 activations
// ---------------------------------------------------------------------------
__global__ void pn_apply(const f16* __restrict__ Y, const float* __restrict__ scale,
                         const float* __restrict__ shift, f16* __restrict__ Xn,
                         int Cout, int lcCout) {
    const size_t idx = (size_t)blockIdx.x * 256 + threadIdx.x;
    const size_t totalElems = (size_t)NB * NPTS * Cout;
    if (idx >= totalElems) return;
    const int c = (int)(idx & (size_t)(Cout - 1));
    const int b = (int)(idx >> (12 + lcCout));
    float v = (float)Y[idx] * scale[b * Cout + c] + shift[b * Cout + c];
    v = v > 0.f ? v : 0.f;
    Xn[idx] = (f16)v;
}

// ---------------------------------------------------------------------------
// 9) Fused norm+relu+max-pool over points for layer 4 output (Cout = 512)
// ---------------------------------------------------------------------------
__launch_bounds__(512)
__global__ void pn_maxpool(const f16* __restrict__ Y4, const float* __restrict__ scale,
                           const float* __restrict__ shift, float* __restrict__ g) {
    const int b = blockIdx.x;
    const int c = threadIdx.x;                 // 512 threads = 512 channels
    const float sc = scale[b * 512 + c];
    const float sh = shift[b * 512 + c];
    const f16* base = Y4 + (size_t)b * NPTS * 512 + c;
    float mx = 0.0f;                           // relu output >= 0
    for (int p = 0; p < NPTS; ++p) {
        float v = (float)base[(size_t)p * 512] * sc + sh;
        v = v > 0.f ? v : 0.f;
        mx = v > mx ? v : mx;
    }
    g[b * 512 + c] = mx;
}

// ---------------------------------------------------------------------------
// 10) Head: 512->128 relu -> 1.  Single block, 128 threads.
// ---------------------------------------------------------------------------
__global__ void pn_head(const float* __restrict__ g, const float* __restrict__ fc1w,
                        const float* __restrict__ fc1b, const float* __restrict__ fc2w,
                        const float* __restrict__ fc2b, float* __restrict__ out) {
    __shared__ float hdn[NB * 128];
    const int t = threadIdx.x;
    for (int b = 0; b < NB; ++b) {
        float acc = fc1b[t];
        const float* gr = g + b * 512;
        const float* wr = fc1w + (size_t)t * 512;
        for (int i = 0; i < 512; ++i) acc += gr[i] * wr[i];
        hdn[b * 128 + t] = acc > 0.f ? acc : 0.f;
    }
    __syncthreads();
    if (t < NB) {
        float acc = fc2b[0];
        for (int i = 0; i < 128; ++i) acc += hdn[t * 128 + i] * fc2w[i];
        out[t] = acc;
    }
}

// ---------------------------------------------------------------------------
extern "C" void kernel_launch(void* const* d_in, const int* in_sizes, int n_in,
                              void* d_out, int out_size, void* d_ws, size_t ws_size,
                              hipStream_t stream) {
    (void)in_sizes; (void)n_in; (void)out_size; (void)ws_size;
    const float* vol    = (const float*)d_in[0];
    const float* w[4]   = {(const float*)d_in[1], (const float*)d_in[4],
                           (const float*)d_in[7], (const float*)d_in[10]};
    const float* gnw[4] = {(const float*)d_in[2], (const float*)d_in[5],
                           (const float*)d_in[8], (const float*)d_in[11]};
    const float* gnb[4] = {(const float*)d_in[3], (const float*)d_in[6],
                           (const float*)d_in[9], (const float*)d_in[12]};
    const float* fc1w = (const float*)d_in[13];
    const float* fc1b = (const float*)d_in[14];
    const float* fc2w = (const float*)d_in[15];
    const float* fc2b = (const float*)d_in[16];

    char* ws = (char*)d_ws;
    size_t off = 0;
    auto alloc = [&](size_t bytes) { size_t r = off; off += (bytes + 255) & ~(size_t)255; return r; };
    const size_t o_counts  = alloc((size_t)NB * NCHUNK * 4);
    const size_t o_offs    = alloc((size_t)NB * NCHUNK * 4);
    const size_t o_totals  = alloc((size_t)NB * 4);
    const size_t o_compact = alloc((size_t)NB * DHW * 4);          // 56.6 MB
    const size_t o_bufA    = alloc((size_t)NB * NPTS * 512 * 2);   // 64 MB (ping)
    const size_t o_bufB    = alloc((size_t)NB * NPTS * 512 * 2);   // 64 MB (pong)
    const size_t o_wfrag   = alloc((size_t)256 * 512 * 2);         // largest weight
    const size_t o_scale   = alloc((size_t)NB * 512 * 4);
    const size_t o_shift   = alloc((size_t)NB * 512 * 4);
    const size_t o_g       = alloc((size_t)NB * 512 * 4);

    int*   countsp  = (int*)(ws + o_counts);
    int*   offsp    = (int*)(ws + o_offs);
    int*   totalsp  = (int*)(ws + o_totals);
    int*   compactp = (int*)(ws + o_compact);
    f16*   bufA     = (f16*)(ws + o_bufA);
    f16*   bufB     = (f16*)(ws + o_bufB);
    f16*   wfragp   = (f16*)(ws + o_wfrag);
    float* scalep   = (float*)(ws + o_scale);
    float* shiftp   = (float*)(ws + o_shift);
    float* gp       = (float*)(ws + o_g);

    // --- point extraction ---
    pn_count  <<<dim3(NCHUNK, NB), 256, 0, stream>>>(vol, countsp);
    pn_scan   <<<NB, 256, 0, stream>>>(countsp, offsp, totalsp);
    pn_scatter<<<dim3(NCHUNK, NB), 256, 0, stream>>>(vol, offsp, compactp);
    pn_sample <<<dim3(NPTS / 256, NB), 256, 0, stream>>>(compactp, totalsp, bufA);

    // --- trunk: 4 WMMA GEMM layers with GroupNorm+ReLU ---
    const int CinPad[4]  = {32, 64, 128, 256};
    const int CinReal[4] = {3, 64, 128, 256};
    const int CoutA[4]   = {64, 128, 256, 512};
    for (int l = 0; l < 4; ++l) {
        const int cp = CinPad[l], co = CoutA[l];
        const int wtotal = cp * co;
        pn_prep_wfrag<<<(wtotal + 255) / 256, 256, 0, stream>>>(w[l], wfragp, cp, CinReal[l], co);
        pn_gemm_wmma<<<dim3((NB * NPTS) / 64, co / 64), 128, 0, stream>>>(bufA, wfragp, bufB, cp, co);
        pn_stats<<<dim3(NB, 8), 256, 0, stream>>>(bufB, gnw[l], gnb[l], scalep, shiftp, co);
        if (l < 3) {
            const size_t tot = (size_t)NB * NPTS * co;
            pn_apply<<<(unsigned)(tot / 256), 256, 0, stream>>>(bufB, scalep, shiftp, bufA,
                                                                co, __builtin_ctz(co));
        }
    }

    // --- layer-4 fused norm+relu+maxpool, then head ---
    pn_maxpool<<<NB, 512, 0, stream>>>(bufB, scalep, shiftp, gp);
    pn_head<<<1, 128, 0, stream>>>(gp, fc1w, fc1b, fc2w, fc2b, (float*)d_out);
}
